// LocalAttnLayer_33552284517098
// MI455X (gfx1250) — matrix operand order repaired and checked by reading, not compiled, not run on hardware
//
#include <hip/hip_runtime.h>
#include <cstdint>

typedef __attribute__((ext_vector_type(16))) _Float16 v16h;
typedef __attribute__((ext_vector_type(8)))  float    v8f;

union Frag16 { uint32_t u[8]; v16h h; };

__device__ __forceinline__ float gelu_exact(float v) {
    return 0.5f * v * (1.0f + erff(v * 0.70710678118654752440f));
}

// ---------------------------------------------------------------------------
// constants
// ---------------------------------------------------------------------------
constexpr int E    = 192;
constexpr int HEADS= 6;
constexpr int HD   = 32;     // head dim
constexpr int LW   = 64;     // tokens per window (P*P)
constexpr int NW   = 2048;   // windows (8 * 16 * 16)
constexpr int MTOK = NW * LW;       // 131072 token rows
constexpr int HW   = 128;           // spatial H = W

// ---------------------------------------------------------------------------
// weight f32 -> f16 conversion
// ---------------------------------------------------------------------------
__global__ __launch_bounds__(256) void cvt_kernel(const float* __restrict__ src,
                                                  _Float16* __restrict__ dst, int n) {
    int i = blockIdx.x * 256 + threadIdx.x;
    if (i < n) dst[i] = (_Float16)src[i];
}

// ---------------------------------------------------------------------------
// rearrange b c (hp p1) (wp p2) -> (win, l, c) and add positional embedding
// ---------------------------------------------------------------------------
__global__ __launch_bounds__(256) void prep_kernel(const float* __restrict__ x,
                                                   const float* __restrict__ row_e,
                                                   const float* __restrict__ col_e,
                                                   float* __restrict__ resid) {
    size_t idx = (size_t)blockIdx.x * 256 + threadIdx.x;  // (b, c, h, w) order
    int w = idx & 127;
    size_t t = idx >> 7;
    int h = t & 127; t >>= 7;
    int c = (int)(t % E);
    int b = (int)(t / E);
    int hp = h >> 3, p1 = h & 7, wp = w >> 3, p2 = w & 7;
    int win = (b << 8) + (hp << 4) + wp;
    int l   = (p1 << 3) + p2;
    float pos = (c < 96) ? row_e[p1 * 96 + c] : col_e[p2 * 96 + (c - 96)];
    resid[((size_t)win * LW + l) * E + c] = x[idx] + pos;
}

// ---------------------------------------------------------------------------
// layernorm over 192 features, one wave (32 lanes) per row, f16 output
// ---------------------------------------------------------------------------
__global__ __launch_bounds__(256) void ln_kernel(const float* __restrict__ in,
                                                 const float* __restrict__ g,
                                                 const float* __restrict__ b,
                                                 _Float16* __restrict__ out) {
    int wid = threadIdx.x >> 5, lane = threadIdx.x & 31;
    int row = blockIdx.x * 8 + wid;
    const float* x = in + (size_t)row * E;
    float v[6], s = 0.f;
#pragma unroll
    for (int j = 0; j < 6; j++) { v[j] = x[lane + 32 * j]; s += v[j]; }
#pragma unroll
    for (int m = 16; m >= 1; m >>= 1) s += __shfl_xor(s, m, 32);
    float mean = s * (1.f / E);
    float s2 = 0.f;
#pragma unroll
    for (int j = 0; j < 6; j++) { float d = v[j] - mean; s2 += d * d; }
#pragma unroll
    for (int m = 16; m >= 1; m >>= 1) s2 += __shfl_xor(s2, m, 32);
    float rstd = rsqrtf(s2 * (1.f / E) + 1e-5f);
#pragma unroll
    for (int j = 0; j < 6; j++) {
        int c = lane + 32 * j;
        out[(size_t)row * E + c] = (_Float16)((v[j] - mean) * rstd * g[c] + b[c]);
    }
}

// ---------------------------------------------------------------------------
// generic GEMM: Y[M,N] = A[M,K] @ W[N,K]^T + bias, f16 inputs, f32 accumulate
// block tile 128x64, 8 waves (4x2), each wave 32x32 (2x2 wmma tiles),
// compile-time K/N, double-buffered LDS staging.
// MODE 0: +bias                       -> f16   (QKV)
// MODE 1: +bias +resid                -> f32   (out-proj -> resid2)
// MODE 2: +bias, GELU                 -> f16   (FFN1)
// MODE 3: +bias, GELU, +resid, scatter-> f32 BCHW (FFN2 -> d_out)
// ---------------------------------------------------------------------------
template <int MODE, int K, int N>
__global__ __launch_bounds__(256) void gemm_kernel(
    const _Float16* __restrict__ A, const _Float16* __restrict__ W,
    const float* __restrict__ bias,
    const float* __restrict__ resid, float* __restrict__ out_f32,
    _Float16* __restrict__ out_h16) {
    constexpr int KT = K / 32;
    __shared__ __align__(16) _Float16 As[2][128 * 36];
    __shared__ __align__(16) _Float16 Bs[2][64 * 36];

    const int tid  = threadIdx.x;
    const int wid  = tid >> 5;
    const int lane = tid & 31;
    const int wm   = wid & 3, wn = wid >> 2;
    const int m0   = blockIdx.x * 128;
    const int n0   = blockIdx.y * 64;
    const int nl   = lane & 15, g = lane >> 4;

    const int arow = tid >> 1, akh = (tid & 1) * 16;   // A: 128 rows x 32 halves
    const int brow = tid >> 2, bkh = (tid & 3) * 8;    // B: 64 rows x 32 halves
    const _Float16* aptr = A + (size_t)(m0 + arow) * K + akh;
    const _Float16* bptr = W + (size_t)(n0 + brow) * K + bkh;

    auto stage = [&](int buf, int k0) {
        uint4 d0 = *(const uint4*)(aptr + k0);
        uint4 d1 = *(const uint4*)(aptr + k0 + 8);
        uint32_t* ad = (uint32_t*)&As[buf][arow * 36 + akh];
        ad[0] = d0.x; ad[1] = d0.y; ad[2] = d0.z; ad[3] = d0.w;
        ad[4] = d1.x; ad[5] = d1.y; ad[6] = d1.z; ad[7] = d1.w;
        uint4 db = *(const uint4*)(bptr + k0);
        uint32_t* bd = (uint32_t*)&Bs[buf][brow * 36 + bkh];
        bd[0] = db.x; bd[1] = db.y; bd[2] = db.z; bd[3] = db.w;
    };

    v8f acc[2][2];
#pragma unroll
    for (int i = 0; i < 2; i++)
#pragma unroll
        for (int j = 0; j < 2; j++)
#pragma unroll
            for (int e = 0; e < 8; e++) acc[i][j][e] = 0.f;

    stage(0, 0);
    __syncthreads();

#pragma unroll
    for (int kt = 0; kt < KT; kt++) {
        const int buf = kt & 1;
        if (kt + 1 < KT) stage(buf ^ 1, (kt + 1) * 32);   // prefetch next tile

        Frag16 a[2], b[2];
#pragma unroll
        for (int am = 0; am < 2; am++) {
            int m = 32 * wm + 16 * am + nl;
#pragma unroll
            for (int v = 0; v < 8; v++) {
                int kp = (v < 4) ? (8 * g + 2 * v) : (8 * g + 2 * v + 8);
                a[am].u[v] = *(const uint32_t*)&As[buf][m * 36 + kp];
            }
        }
#pragma unroll
        for (int an = 0; an < 2; an++) {
            int n = 32 * wn + 16 * an + nl;
#pragma unroll
            for (int v = 0; v < 8; v++) {
                int kp = (v < 4) ? (8 * g + 2 * v) : (8 * g + 2 * v + 8);
                b[an].u[v] = *(const uint32_t*)&Bs[buf][n * 36 + kp];
            }
        }
#pragma unroll
        for (int am = 0; am < 2; am++)
#pragma unroll
            for (int an = 0; an < 2; an++)
                acc[am][an] = __builtin_amdgcn_wmma_f32_16x16x32_f16(
                    false, a[am].h, false, b[an].h, (short)0, acc[am][an], false, false);
        __syncthreads();
    }

    // epilogue: lane holds D[8*g+v][nl] of each 16x16 tile
#pragma unroll
    for (int am = 0; am < 2; am++)
#pragma unroll
        for (int an = 0; an < 2; an++) {
            int col = n0 + 32 * wn + 16 * an + nl;
            float bv = bias[col];
#pragma unroll
            for (int v = 0; v < 8; v++) {
                int m = m0 + 32 * wm + 16 * am + 8 * g + v;
                float val = acc[am][an][v] + bv;
                if (MODE == 0) {
                    out_h16[(size_t)m * N + col] = (_Float16)val;
                } else if (MODE == 1) {
                    val += resid[(size_t)m * E + col];
                    out_f32[(size_t)m * E + col] = val;
                } else if (MODE == 2) {
                    out_h16[(size_t)m * N + col] = (_Float16)gelu_exact(val);
                } else {
                    val = gelu_exact(val) + resid[(size_t)m * E + col];
                    int win = m >> 6, l = m & 63;
                    int bb = win >> 8, rem = win & 255;
                    int hp = rem >> 4, wp = rem & 15;
                    int hh = hp * 8 + (l >> 3), ww = wp * 8 + (l & 7);
                    out_f32[(((size_t)bb * E + col) * HW + hh) * HW + ww] = val;
                }
            }
        }
}

// ---------------------------------------------------------------------------
// attention: one block per (window, head); 4 waves, 64x64 scores, softmax,
// O = P @ V. All matmuls via v_wmma_f32_16x16x32_f16 (HD = 32 = one K step).
// ---------------------------------------------------------------------------
__global__ __launch_bounds__(128) void attn_kernel(const _Float16* __restrict__ qkv,
                                                   _Float16* __restrict__ o) {
    __shared__ __align__(16) _Float16 Qs[64 * 36];   // [l][d]
    __shared__ __align__(16) _Float16 Ks[64 * 36];   // [l][d]
    __shared__ __align__(16) _Float16 Vst[32 * 68];  // [d][l]  (transposed)
    __shared__ __align__(16) _Float16 Ps[64 * 68];   // [i][j]  softmax probs

    const int win  = blockIdx.x;
    const int head = blockIdx.y;
    const int tid  = threadIdx.x;
    const size_t base = (size_t)win * LW * (3 * E) + head * HD;

    {   // stage Q, K, V(transposed): thread -> (row l, half kh)
        int l  = tid >> 1;
        int kh = (tid & 1) * 16;
        const _Float16* q = qkv + base + (size_t)l * (3 * E) + kh;
        uint4 q0 = *(const uint4*)q;       uint4 q1 = *(const uint4*)(q + 8);
        uint4 k0 = *(const uint4*)(q + E); uint4 k1 = *(const uint4*)(q + E + 8);
        uint32_t* qd = (uint32_t*)&Qs[l * 36 + kh];
        qd[0]=q0.x; qd[1]=q0.y; qd[2]=q0.z; qd[3]=q0.w;
        qd[4]=q1.x; qd[5]=q1.y; qd[6]=q1.z; qd[7]=q1.w;
        uint32_t* kd = (uint32_t*)&Ks[l * 36 + kh];
        kd[0]=k0.x; kd[1]=k0.y; kd[2]=k0.z; kd[3]=k0.w;
        kd[4]=k1.x; kd[5]=k1.y; kd[6]=k1.z; kd[7]=k1.w;
        _Float16 vtmp[16];
        *(uint4*)&vtmp[0] = *(const uint4*)(q + 2 * E);
        *(uint4*)&vtmp[8] = *(const uint4*)(q + 2 * E + 8);
#pragma unroll
        for (int i = 0; i < 16; i++) Vst[(kh + i) * 68 + l] = vtmp[i];
    }
    __syncthreads();

    const int wid = tid >> 5, lane = tid & 31;
    const int i0 = wid * 16;           // this wave's score rows
    const int nl = lane & 15, g = lane >> 4;

    // A-fragment of Q rows [i0, i0+16), K dim = HD = 32
    Frag16 aq;
#pragma unroll
    for (int v = 0; v < 8; v++) {
        int kp = (v < 4) ? (8 * g + 2 * v) : (8 * g + 2 * v + 8);
        aq.u[v] = *(const uint32_t*)&Qs[(i0 + nl) * 36 + kp];
    }

    // S = Q K^T : 4 column tiles of 16
    v8f sc[4];
#pragma unroll
    for (int jt = 0; jt < 4; jt++) {
        Frag16 bk;   // B[d][j] = K[j][d]; lane = column j
#pragma unroll
        for (int v = 0; v < 8; v++) {
            int kp = (v < 4) ? (8 * g + 2 * v) : (8 * g + 2 * v + 8);
            bk.u[v] = *(const uint32_t*)&Ks[(jt * 16 + nl) * 36 + kp];
        }
        v8f z;
#pragma unroll
        for (int e = 0; e < 8; e++) z[e] = 0.f;
        sc[jt] = __builtin_amdgcn_wmma_f32_16x16x32_f16(false, aq.h, false, bk.h,
                                                        (short)0, z, false, false);
    }

    // fp32 softmax over j (64 values per row), rows = i0 + 8*g + v
    const float scale = 0.17677669529663687f;  // 1/sqrt(32)
#pragma unroll
    for (int v = 0; v < 8; v++) {
        float mx = -1e30f;
#pragma unroll
        for (int jt = 0; jt < 4; jt++) { sc[jt][v] *= scale; mx = fmaxf(mx, sc[jt][v]); }
#pragma unroll
        for (int m = 1; m < 16; m <<= 1) mx = fmaxf(mx, __shfl_xor(mx, m, 32));
        float sum = 0.f;
#pragma unroll
        for (int jt = 0; jt < 4; jt++) { float p = expf(sc[jt][v] - mx); sc[jt][v] = p; sum += p; }
#pragma unroll
        for (int m = 1; m < 16; m <<= 1) sum += __shfl_xor(sum, m, 32);
        float inv = 1.f / sum;
#pragma unroll
        for (int jt = 0; jt < 4; jt++)
            Ps[(i0 + 8 * g + v) * 68 + jt * 16 + nl] = (_Float16)(sc[jt][v] * inv);
    }
    __syncthreads();

    // O = P @ V : K dim 64 -> two K=32 steps; N dim 32 -> two 16-col tiles
    v8f ov[2];
#pragma unroll
    for (int dt = 0; dt < 2; dt++)
#pragma unroll
        for (int e = 0; e < 8; e++) ov[dt][e] = 0.f;

#pragma unroll
    for (int j0 = 0; j0 < 64; j0 += 32) {
        Frag16 ap;
#pragma unroll
        for (int v = 0; v < 8; v++) {
            int kp = (v < 4) ? (8 * g + 2 * v) : (8 * g + 2 * v + 8);
            ap.u[v] = *(const uint32_t*)&Ps[(i0 + nl) * 68 + j0 + kp];
        }
#pragma unroll
        for (int dt = 0; dt < 2; dt++) {
            Frag16 bv;   // B[k][n] = V[j0+k][dt*16+n] = Vst[dt*16+n][j0+k]
#pragma unroll
            for (int v = 0; v < 8; v++) {
                int kp = (v < 4) ? (8 * g + 2 * v) : (8 * g + 2 * v + 8);
                bv.u[v] = *(const uint32_t*)&Vst[(dt * 16 + nl) * 68 + j0 + kp];
            }
            ov[dt] = __builtin_amdgcn_wmma_f32_16x16x32_f16(false, ap.h, false, bv.h,
                                                            (short)0, ov[dt], false, false);
        }
    }

#pragma unroll
    for (int dt = 0; dt < 2; dt++)
#pragma unroll
        for (int v = 0; v < 8; v++) {
            int i = i0 + 8 * g + v;
            int d = dt * 16 + nl;
            o[((size_t)win * LW + i) * E + head * HD + d] = (_Float16)ov[dt][v];
        }
}

// ---------------------------------------------------------------------------
// host launch
// ---------------------------------------------------------------------------
extern "C" void kernel_launch(void* const* d_in, const int* in_sizes, int n_in,
                              void* d_out, int out_size, void* d_ws, size_t ws_size,
                              hipStream_t stream) {
    const float* x     = (const float*)d_in[0];
    const float* row_e = (const float*)d_in[1];
    const float* col_e = (const float*)d_in[2];
    const float* ln1g  = (const float*)d_in[3];
    const float* ln1b  = (const float*)d_in[4];
    const float* wqkv  = (const float*)d_in[5];
    const float* bqkv  = (const float*)d_in[6];
    const float* wout  = (const float*)d_in[7];
    const float* bout  = (const float*)d_in[8];
    const float* ln2g  = (const float*)d_in[9];
    const float* ln2b  = (const float*)d_in[10];
    const float* wf1   = (const float*)d_in[11];
    const float* bf1   = (const float*)d_in[12];
    const float* wf2   = (const float*)d_in[13];
    const float* bf2   = (const float*)d_in[14];
    float* out = (float*)d_out;

    char* ws = (char*)d_ws;
    size_t off = 0;
    auto alloc = [&](size_t bytes) -> void* {
        void* p = ws + off;
        off += (bytes + 255) & ~(size_t)255;
        return p;
    };
    _Float16* w_qkv_h = (_Float16*)alloc((size_t)3 * E * E * 2);       // 576x192
    _Float16* w_out_h = (_Float16*)alloc((size_t)E * E * 2);           // 192x192
    _Float16* w_f1_h  = (_Float16*)alloc((size_t)4 * E * E * 2);       // 768x192
    _Float16* w_f2_h  = (_Float16*)alloc((size_t)E * 4 * E * 2);       // 192x768
    float*    resid   = (float*)alloc((size_t)MTOK * E * 4);           // 100.7 MB
    _Float16* bufA    = (_Float16*)alloc((size_t)MTOK * E * 2);        // xn / o / ln2-out
    char*     bufB    = (char*)alloc((size_t)MTOK * 3 * E * 2);        // qkv, later resid2
    _Float16* bufC    = (_Float16*)alloc((size_t)MTOK * 4 * E * 2);    // ffn hidden

    _Float16* qkv_h  = (_Float16*)bufB;
    float*    resid2 = (float*)bufB;   // reuse after attention consumes qkv

    // 1. weights -> f16
    cvt_kernel<<<(3 * E * E + 255) / 256, 256, 0, stream>>>(wqkv, w_qkv_h, 3 * E * E);
    cvt_kernel<<<(E * E + 255) / 256, 256, 0, stream>>>(wout, w_out_h, E * E);
    cvt_kernel<<<(4 * E * E + 255) / 256, 256, 0, stream>>>(wf1, w_f1_h, 4 * E * E);
    cvt_kernel<<<(4 * E * E + 255) / 256, 256, 0, stream>>>(wf2, w_f2_h, 4 * E * E);

    // 2. rearrange + positional embedding -> resid (fp32)
    prep_kernel<<<(size_t)MTOK * E / 256, 256, 0, stream>>>(x, row_e, col_e, resid);

    // 3. LN1 -> f16
    ln_kernel<<<MTOK / 8, 256, 0, stream>>>(resid, ln1g, ln1b, bufA);

    // 4. QKV projection: (M,192) @ (192,576)
    gemm_kernel<0, 192, 576><<<dim3(MTOK / 128, 9), 256, 0, stream>>>(
        bufA, w_qkv_h, bqkv, nullptr, nullptr, qkv_h);

    // 5. windowed attention -> bufA (f16, head-concat features)
    attn_kernel<<<dim3(NW, HEADS), 128, 0, stream>>>(qkv_h, bufA);

    // 6. output projection + residual -> resid2 (fp32)
    gemm_kernel<1, 192, 192><<<dim3(MTOK / 128, 3), 256, 0, stream>>>(
        bufA, w_out_h, bout, resid, resid2, nullptr);

    // 7. LN2 -> f16
    ln_kernel<<<MTOK / 8, 256, 0, stream>>>(resid2, ln2g, ln2b, bufA);

    // 8. FFN1 + GELU -> f16 hidden
    gemm_kernel<2, 192, 768><<<dim3(MTOK / 128, 12), 256, 0, stream>>>(
        bufA, w_f1_h, bf1, nullptr, nullptr, bufC);

    // 9. FFN2 + GELU + residual, scattered directly into BCHW output
    gemm_kernel<3, 768, 192><<<dim3(MTOK / 128, 3), 256, 0, stream>>>(
        bufC, w_f2_h, bf2, resid2, out, nullptr);
}
